// OptionEncoder_MLP_76665166234145
// MI455X (gfx1250) — matrix operand order repaired and checked by reading, not compiled, not run on hardware
//
#include <hip/hip_runtime.h>

// OptionEncoder gumbel-softmax scan, forward/eval:
//   per step: q = relu(relu([s|c]W1+b1)W2+b2)W3+b3 ; c' = onehot(argmax(q+g)) per group of 8
// (softmax/tau drop out of the hard straight-through forward path.)

typedef __attribute__((ext_vector_type(16))) _Float16 v16h;
typedef __attribute__((ext_vector_type(8)))  _Float16 v8h;
typedef __attribute__((ext_vector_type(8)))  float    v8f;
typedef __attribute__((ext_vector_type(4)))  float    v4f;

#define NB   4096   // batch
#define NT   256    // timesteps
#define ND   64     // state dim
#define NH   32     // mlp hidden
#define NOUT 32     // LD*CD

static __device__ __forceinline__ v8f wmma_f16(v16h a, v16h b, v8f c) {
  // (neg_a, A, neg_b, B, c_mod, C, reuse_a, reuse_b)
  return __builtin_amdgcn_wmma_f32_16x16x32_f16(false, a, false, b, (short)0, c,
                                                false, false);
}

__global__ __launch_bounds__(32) void option_encoder_scan_kernel(
    const float* __restrict__ states,   // [B,T,64] f32
    const float* __restrict__ gumbel,   // [B,T,32] f32
    const float* __restrict__ W1, const float* __restrict__ b1,  // [96,32],[32]
    const float* __restrict__ W2, const float* __restrict__ b2,  // [32,32],[32]
    const float* __restrict__ W3, const float* __restrict__ b3,  // [32,32],[32]
    float* __restrict__ out)            // [B,T,32] f32 (hard one-hot)
{
  __shared__ alignas(16) _Float16 hbuf[16 * NH];    // h1/h2 staging (f16, [M][N])
  __shared__ alignas(16) _Float16 cbuf[16 * NOUT];  // recurrent one-hot state (f16)
  __shared__ alignas(16) float    qbuf[16 * NOUT];  // fc3 output (f32)

  const int lane = threadIdx.x & 31;
  const int lrow = lane & 15;   // M for A-layout, N for B/C/D-layout
  const int lhi  = lane >> 4;   // lane half
  const int b0   = blockIdx.x * 16;

  // ---------------- hoist weights into WMMA B-operand registers ----------------
  // B layout (f16 32x16 KxN): lane -> N = lane&15 ; half slot h -> K = h + (h&8) + 8*(lane>>4)
  v16h Bw1[3][2], Bw2[2], Bw3[2];
#pragma unroll
  for (int kc = 0; kc < 3; ++kc) {
#pragma unroll
    for (int nt = 0; nt < 2; ++nt) {
      v16h bb;
      const int n  = nt * 16 + lrow;
      const int kb = kc * 32 + lhi * 8;
#pragma unroll
      for (int h = 0; h < 16; ++h) bb[h] = (_Float16)W1[(kb + h + (h & 8)) * NOUT + n];
      Bw1[kc][nt] = bb;
    }
  }
#pragma unroll
  for (int nt = 0; nt < 2; ++nt) {
    v16h b2v, b3v;
    const int n  = nt * 16 + lrow;
    const int kb = lhi * 8;
#pragma unroll
    for (int h = 0; h < 16; ++h) {
      b2v[h] = (_Float16)W2[(kb + h + (h & 8)) * NH + n];
      b3v[h] = (_Float16)W3[(kb + h + (h & 8)) * NH + n];
    }
    Bw2[nt] = b2v; Bw3[nt] = b3v;
  }
  // biases broadcast into accumulator-init (C/D layout: N fixed per lane)
  float bias1[2], bias2[2], bias3[2];
#pragma unroll
  for (int nt = 0; nt < 2; ++nt) {
    bias1[nt] = b1[nt * 16 + lrow];
    bias2[nt] = b2[nt * 16 + lrow];
    bias3[nt] = b3[nt * 16 + lrow];
  }

  // c0 = one-hot(index 0) per latent group of 8
  for (int i = lane; i < 16 * NOUT; i += 32)
    cbuf[i] = (_Float16)(((i & 7) == 0) ? 1.0f : 0.0f);
  __syncthreads();

  // ---------------- software-pipelined states loads ----------------
  // A layout (f16 16x32 MxK): lane -> M = lane&15 ;
  // slots 0..7: K = kbase + 8*lhi + j ; slots 8..15: K = kbase + 16 + 8*lhi + j
  const float* srow0 = states + (size_t)(b0 + lrow) * NT * ND;
  v4f sreg[8];
#pragma unroll
  for (int kc = 0; kc < 2; ++kc) {
    sreg[kc * 4 + 0] = *(const v4f*)(srow0 + kc * 32 + lhi * 8);
    sreg[kc * 4 + 1] = *(const v4f*)(srow0 + kc * 32 + lhi * 8 + 4);
    sreg[kc * 4 + 2] = *(const v4f*)(srow0 + kc * 32 + 16 + lhi * 8);
    sreg[kc * 4 + 3] = *(const v4f*)(srow0 + kc * 32 + 16 + lhi * 8 + 4);
  }

  // per-lane argmax task geometry (2 tasks/lane): row/lat fixed across t
  int trow[2], tlat[2];
#pragma unroll
  for (int tk0 = 0; tk0 < 2; ++tk0) {
    const int tk = lane + 32 * tk0;
    trow[tk0] = tk >> 2;
    tlat[tk0] = tk & 3;
  }

  for (int t = 0; t < NT; ++t) {
    // ---------- convert current states regs -> f16 A operands ----------
    v16h a01[2];
#pragma unroll
    for (int kc = 0; kc < 2; ++kc) {
      v16h av;
#pragma unroll
      for (int j = 0; j < 4; ++j) {
        av[j]      = (_Float16)sreg[kc * 4 + 0][j];
        av[4 + j]  = (_Float16)sreg[kc * 4 + 1][j];
        av[8 + j]  = (_Float16)sreg[kc * 4 + 2][j];
        av[12 + j] = (_Float16)sreg[kc * 4 + 3][j];
      }
      a01[kc] = av;
    }

    // ---------- prefetch states for t+1 (hidden behind the 3 GEMMs) ----------
    {
      const int tn = (t + 1 < NT) ? (t + 1) : t;   // last iter: harmless reload
      const float* srow = srow0 + (size_t)tn * ND;
#pragma unroll
      for (int kc = 0; kc < 2; ++kc) {
        sreg[kc * 4 + 0] = *(const v4f*)(srow + kc * 32 + lhi * 8);
        sreg[kc * 4 + 1] = *(const v4f*)(srow + kc * 32 + lhi * 8 + 4);
        sreg[kc * 4 + 2] = *(const v4f*)(srow + kc * 32 + 16 + lhi * 8);
        sreg[kc * 4 + 3] = *(const v4f*)(srow + kc * 32 + 16 + lhi * 8 + 4);
      }
    }

    // ---------- prefetch gumbel for this step (consumed after fc3) ----------
    v4f greg[2][2];
    size_t gofs[2];
#pragma unroll
    for (int tk0 = 0; tk0 < 2; ++tk0) {
      gofs[tk0] = ((size_t)(b0 + trow[tk0]) * NT + t) * NOUT + tlat[tk0] * 8;
      greg[tk0][0] = *(const v4f*)(gumbel + gofs[tk0]);
      greg[tk0][1] = *(const v4f*)(gumbel + gofs[tk0] + 4);
    }

    // ---------- recurrent one-hot chunk (K = 64..95 of the concat) ----------
    v16h a2;
    {
      const _Float16* crow = cbuf + lrow * NOUT;
      const v8h c0 = *(const v8h*)(crow + lhi * 8);
      const v8h c1 = *(const v8h*)(crow + 16 + lhi * 8);
#pragma unroll
      for (int j = 0; j < 8; ++j) { a2[j] = c0[j]; a2[8 + j] = c1[j]; }
    }

    // ---------- fc1 (K=96) + ReLU ----------
    v8f acc1[2];
#pragma unroll
    for (int nt = 0; nt < 2; ++nt) {
      v8f c;
#pragma unroll
      for (int r = 0; r < 8; ++r) c[r] = bias1[nt];
      c = wmma_f16(a01[0], Bw1[0][nt], c);
      c = wmma_f16(a01[1], Bw1[1][nt], c);
      c = wmma_f16(a2,     Bw1[2][nt], c);
      acc1[nt] = c;
    }
    __syncthreads();
#pragma unroll
    for (int nt = 0; nt < 2; ++nt)
#pragma unroll
      for (int r = 0; r < 8; ++r)
        hbuf[(r + 8 * lhi) * NH + nt * 16 + lrow] =
            (_Float16)fmaxf(acc1[nt][r], 0.0f);
    __syncthreads();
    v16h ah;
    {
      const _Float16* hr = hbuf + lrow * NH;
      const v8h h0 = *(const v8h*)(hr + lhi * 8);
      const v8h h1 = *(const v8h*)(hr + 16 + lhi * 8);
#pragma unroll
      for (int j = 0; j < 8; ++j) { ah[j] = h0[j]; ah[8 + j] = h1[j]; }
    }

    // ---------- fc2 (K=32) + ReLU ----------
    v8f acc2[2];
#pragma unroll
    for (int nt = 0; nt < 2; ++nt) {
      v8f c;
#pragma unroll
      for (int r = 0; r < 8; ++r) c[r] = bias2[nt];
      acc2[nt] = wmma_f16(ah, Bw2[nt], c);
    }
    __syncthreads();
#pragma unroll
    for (int nt = 0; nt < 2; ++nt)
#pragma unroll
      for (int r = 0; r < 8; ++r)
        hbuf[(r + 8 * lhi) * NH + nt * 16 + lrow] =
            (_Float16)fmaxf(acc2[nt][r], 0.0f);
    __syncthreads();
    v16h ah2;
    {
      const _Float16* hr = hbuf + lrow * NH;
      const v8h h0 = *(const v8h*)(hr + lhi * 8);
      const v8h h1 = *(const v8h*)(hr + 16 + lhi * 8);
#pragma unroll
      for (int j = 0; j < 8; ++j) { ah2[j] = h0[j]; ah2[8 + j] = h1[j]; }
    }

    // ---------- fc3 (K=32) -> q ----------
    v8f acc3[2];
#pragma unroll
    for (int nt = 0; nt < 2; ++nt) {
      v8f c;
#pragma unroll
      for (int r = 0; r < 8; ++r) c[r] = bias3[nt];
      acc3[nt] = wmma_f16(ah2, Bw3[nt], c);
    }
#pragma unroll
    for (int nt = 0; nt < 2; ++nt)
#pragma unroll
      for (int r = 0; r < 8; ++r)
        qbuf[(r + 8 * lhi) * NOUT + nt * 16 + lrow] = acc3[nt][r];
    __syncthreads();

    // ---------- hard gumbel argmax: 64 (row,latent) tasks, 2 per lane ----------
#pragma unroll
    for (int tk0 = 0; tk0 < 2; ++tk0) {
      const float* qp = qbuf + trow[tk0] * NOUT + tlat[tk0] * 8;
      const v4f q0 = *(const v4f*)(qp);
      const v4f q1 = *(const v4f*)(qp + 4);
      float vals[8];
#pragma unroll
      for (int j = 0; j < 4; ++j) {
        vals[j]     = q0[j] + greg[tk0][0][j];
        vals[4 + j] = q1[j] + greg[tk0][1][j];
      }
      int bi = 0; float best = vals[0];
#pragma unroll
      for (int j = 1; j < 8; ++j)
        if (vals[j] > best) { best = vals[j]; bi = j; }   // first-max, like jnp.argmax
      v8h oh; v4f o0, o1;
#pragma unroll
      for (int j = 0; j < 4; ++j) {
        const float a0 = (bi == j)     ? 1.0f : 0.0f;
        const float a1 = (bi == 4 + j) ? 1.0f : 0.0f;
        o0[j] = a0; o1[j] = a1;
        oh[j] = (_Float16)a0; oh[4 + j] = (_Float16)a1;
      }
      *(v8h*)(cbuf + trow[tk0] * NOUT + tlat[tk0] * 8) = oh;  // next-step state
      float* op = out + gofs[tk0];                            // same [B,T,32] index
      *(v4f*)(op)     = o0;
      *(v4f*)(op + 4) = o1;
    }
    __syncthreads();
  }
}

extern "C" void kernel_launch(void* const* d_in, const int* in_sizes, int n_in,
                              void* d_out, int out_size, void* d_ws, size_t ws_size,
                              hipStream_t stream) {
  (void)in_sizes; (void)n_in; (void)out_size; (void)d_ws; (void)ws_size;
  const float* states = (const float*)d_in[0];
  const float* gumbel = (const float*)d_in[1];
  // d_in[2] = tau: argmax((q+g)/tau) == argmax(q+g) for tau>0, and the hard
  // straight-through forward output is exactly the one-hot -> tau is unused.
  const float* W1 = (const float*)d_in[3];
  const float* b1 = (const float*)d_in[4];
  const float* W2 = (const float*)d_in[5];
  const float* b2 = (const float*)d_in[6];
  const float* W3 = (const float*)d_in[7];
  const float* b3 = (const float*)d_in[8];
  float* out = (float*)d_out;

  dim3 grid(NB / 16);   // 256 batch tiles of M=16, one wave each
  dim3 block(32);
  hipLaunchKernelGGL(option_encoder_scan_kernel, grid, block, 0, stream,
                     states, gumbel, W1, b1, W2, b2, W3, b3, out);
}
